// RGCNLinkPredictor_33036888440840
// MI455X (gfx1250) — matrix operand order repaired and checked by reading, not compiled, not run on hardware
//
#include <hip/hip_runtime.h>
#include <hip/hip_bf16.h>

typedef __attribute__((ext_vector_type(16))) _Float16 v16h;
typedef __attribute__((ext_vector_type(8)))  float    v8f;
typedef __attribute__((ext_vector_type(4)))  float    v4f;

#define NN 100000
#define RR 8
#define BB 4
#define DD 64
#define EE 2000000
#define PP 500000

// ---------------------------------------------------------------------------
// native CDNA5 f32 atomic add (no return -> STOREcnt, no CAS-loop fallback)
// ---------------------------------------------------------------------------
__device__ __forceinline__ void atomAddF(float* p, float v) {
    asm volatile("global_atomic_add_f32 %0, %1, off" :: "v"(p), "v"(v) : "memory");
}

// ---------------------------------------------------------------------------
// zero workspace (sums + counts), grid-stride float4 stores
// ---------------------------------------------------------------------------
__global__ void __launch_bounds__(256) zero_kernel(float* __restrict__ p, long n4) {
    long i      = (long)blockIdx.x * blockDim.x + threadIdx.x;
    long stride = (long)gridDim.x * blockDim.x;
    v4f z = {};
    for (; i < n4; i += stride) ((v4f*)p)[i] = z;
}

// ---------------------------------------------------------------------------
// edge scatter: sums[dst, rel, :] += x[src, :]; cnt[dst, rel] += 1
// 16 threads per edge, each does a float4 worth of atomics
// ---------------------------------------------------------------------------
__global__ void __launch_bounds__(256) scatter_kernel(
        const int* __restrict__ ei, const int* __restrict__ et,
        const float* __restrict__ x, float* __restrict__ sums,
        float* __restrict__ cnt) {
    long tid = (long)blockIdx.x * blockDim.x + threadIdx.x;
    if (tid >= (long)EE * 16) return;
    int e = (int)(tid >> 4);
    int t = (int)(tid & 15);
    int src = ei[e];
    int dst = ei[EE + e];          // edge_index is [2, E] row-major
    int r   = et[e];
    v4f v = ((const v4f*)(x + (size_t)src * DD))[t];
    float* srow = sums + ((size_t)dst * RR + r) * DD + t * 4;
    atomAddF(srow + 0, v[0]);
    atomAddF(srow + 1, v[1]);
    atomAddF(srow + 2, v[2]);
    atomAddF(srow + 3, v[3]);
    if (t == 0) atomAddF(cnt + (size_t)dst * RR + r, 1.0f);
}

// ---------------------------------------------------------------------------
// pack RGCN weights into per-lane WMMA B fragments (f16)
// Wcat[k][o]: k<512 -> sum_b comp[r,b]*basis[b,i,o] (r=k/64,i=k%64); else root
// fragment layout: wfrag[((kc*4+nt)*32+lane)*16 + (2v+j)], kB = half*16+2v+j
// ---------------------------------------------------------------------------
__global__ void __launch_bounds__(256) pack_rgcn_w(
        const float* __restrict__ comp, const float* __restrict__ basis,
        const float* __restrict__ root, _Float16* __restrict__ wfrag) {
    int tid = blockIdx.x * blockDim.x + threadIdx.x;
    if (tid >= 18 * 4 * 32) return;
    int lane = tid & 31, nt = (tid >> 5) & 3, kc = tid >> 7;
    int half = lane >> 4, n = lane & 15, o = nt * 16 + n;
    v16h w;
#pragma unroll
    for (int v = 0; v < 8; ++v) {
#pragma unroll
        for (int j = 0; j < 2; ++j) {
            int k = kc * 32 + half * 16 + v * 2 + j;
            float val;
            if (k < RR * DD) {
                int r = k >> 6, i = k & 63;
                val = 0.0f;
#pragma unroll
                for (int b = 0; b < BB; ++b)
                    val += comp[r * BB + b] * basis[((size_t)b * DD + i) * DD + o];
            } else {
                val = root[(size_t)(k - RR * DD) * DD + o];
            }
            w[v * 2 + j] = (_Float16)val;
        }
    }
    ((v16h*)wfrag)[tid] = w;
}

// pack decoder W1 [128][64] into fragments (K=128 -> 4 k-chunks)
__global__ void __launch_bounds__(256) pack_mlp_w(
        const float* __restrict__ W1, _Float16* __restrict__ wfrag) {
    int tid = blockIdx.x * blockDim.x + threadIdx.x;
    if (tid >= 4 * 4 * 32) return;
    int lane = tid & 31, nt = (tid >> 5) & 3, kc = tid >> 7;
    int half = lane >> 4, n = lane & 15, o = nt * 16 + n;
    v16h w;
#pragma unroll
    for (int v = 0; v < 8; ++v)
#pragma unroll
        for (int j = 0; j < 2; ++j) {
            int k = kc * 32 + half * 16 + v * 2 + j;
            w[v * 2 + j] = (_Float16)W1[(size_t)k * DD + o];
        }
    ((v16h*)wfrag)[tid] = w;
}

#define WMMA_F16(A, B, C) \
    __builtin_amdgcn_wmma_f32_16x16x32_f16(false, (A), false, (B), (short)0, (C), false, false)

// ---------------------------------------------------------------------------
// RGCN node transform: out[n,:] = [mean(n) | xin(n)] @ Wcat + bias (+ReLU)
// One wave per 16-node tile, 4 WMMA accumulators (16x64 output), K=576.
// A fragment built on the fly: sums scaled by rcp(max(cnt,1)), converted to f16.
// ---------------------------------------------------------------------------
__global__ void __launch_bounds__(64) rgcn_gemm(
        const float* __restrict__ sums, const float* __restrict__ cnt,
        const float* __restrict__ xin, const _Float16* __restrict__ wfrag,
        const float* __restrict__ bias, float* __restrict__ out, int do_relu) {
    int lane = threadIdx.x & 31;
    int tile = blockIdx.x * 2 + (threadIdx.x >> 5);
    int m = lane & 15, half = lane >> 4;
    int node = tile * 16 + m;

    const float* srow = sums + (size_t)node * (RR * DD);
    const float* xrow = xin + (size_t)node * DD;

    float invc[RR];
#pragma unroll
    for (int r = 0; r < RR; ++r)
        invc[r] = __builtin_amdgcn_rcpf(fmaxf(cnt[(size_t)node * RR + r], 1.0f));

    v8f acc0 = {}, acc1 = {}, acc2 = {}, acc3 = {};
    const v16h* wf = (const v16h*)wfrag;
    int off = half * 16;

#pragma unroll
    for (int kc = 0; kc < 18; ++kc) {
        const float* base;
        float scale;
        if (kc < 16) {
            base = srow + (kc >> 1) * DD + (kc & 1) * 32;
            scale = invc[kc >> 1];
        } else {
            base = xrow + (kc - 16) * 32;
            scale = 1.0f;
        }
        v4f f0 = *(const v4f*)(base + off);
        v4f f1 = *(const v4f*)(base + off + 4);
        v4f f2 = *(const v4f*)(base + off + 8);
        v4f f3 = *(const v4f*)(base + off + 12);
        v16h a;
#pragma unroll
        for (int q = 0; q < 4; ++q) {
            a[q]      = (_Float16)(f0[q] * scale);
            a[4 + q]  = (_Float16)(f1[q] * scale);
            a[8 + q]  = (_Float16)(f2[q] * scale);
            a[12 + q] = (_Float16)(f3[q] * scale);
        }
        v16h b0 = wf[(kc * 4 + 0) * 32 + lane];
        v16h b1 = wf[(kc * 4 + 1) * 32 + lane];
        v16h b2 = wf[(kc * 4 + 2) * 32 + lane];
        v16h b3 = wf[(kc * 4 + 3) * 32 + lane];
        acc0 = WMMA_F16(a, b0, acc0);
        acc1 = WMMA_F16(a, b1, acc1);
        acc2 = WMMA_F16(a, b2, acc2);
        acc3 = WMMA_F16(a, b3, acc3);
    }

    int n = lane & 15;
#pragma unroll
    for (int nt = 0; nt < 4; ++nt) {
        v8f acc = (nt == 0) ? acc0 : (nt == 1) ? acc1 : (nt == 2) ? acc2 : acc3;
        float bb = bias[nt * 16 + n];
#pragma unroll
        for (int v = 0; v < 8; ++v) {
            float val = acc[v] + bb;
            if (do_relu) val = fmaxf(val, 0.0f);
            out[(size_t)(tile * 16 + half * 8 + v) * DD + nt * 16 + n] = val;
        }
    }
}

// ---------------------------------------------------------------------------
// Decoder: logits[p] = relu([z[s]|z[d]] @ W1 + b1) @ W2 + b2
// One wave per 16-pair tile; GEMM via WMMA, fused ReLU + dot + shfl reduce.
// ---------------------------------------------------------------------------
__global__ void __launch_bounds__(64) decode_kernel(
        const int* __restrict__ pairs, const float* __restrict__ z,
        const _Float16* __restrict__ wfrag, const float* __restrict__ b1,
        const float* __restrict__ W2, const float* __restrict__ b2,
        float* __restrict__ out) {
    int lane = threadIdx.x & 31;
    int tile = blockIdx.x * 2 + (threadIdx.x >> 5);
    int m = lane & 15, half = lane >> 4;
    int p = tile * 16 + m;
    int s = pairs[(size_t)p * 2 + 0];
    int d = pairs[(size_t)p * 2 + 1];
    const float* zs = z + (size_t)s * DD;
    const float* zd = z + (size_t)d * DD;

    v8f acc0 = {}, acc1 = {}, acc2 = {}, acc3 = {};
    const v16h* wf = (const v16h*)wfrag;
    int off = half * 16;

#pragma unroll
    for (int kc = 0; kc < 4; ++kc) {
        const float* base = (kc < 2) ? (zs + kc * 32) : (zd + (kc - 2) * 32);
        v4f f0 = *(const v4f*)(base + off);
        v4f f1 = *(const v4f*)(base + off + 4);
        v4f f2 = *(const v4f*)(base + off + 8);
        v4f f3 = *(const v4f*)(base + off + 12);
        v16h a;
#pragma unroll
        for (int q = 0; q < 4; ++q) {
            a[q]      = (_Float16)f0[q];
            a[4 + q]  = (_Float16)f1[q];
            a[8 + q]  = (_Float16)f2[q];
            a[12 + q] = (_Float16)f3[q];
        }
        v16h b0 = wf[(kc * 4 + 0) * 32 + lane];
        v16h b1 = wf[(kc * 4 + 1) * 32 + lane];
        v16h b2 = wf[(kc * 4 + 2) * 32 + lane];
        v16h b3 = wf[(kc * 4 + 3) * 32 + lane];
        acc0 = WMMA_F16(a, b0, acc0);
        acc1 = WMMA_F16(a, b1, acc1);
        acc2 = WMMA_F16(a, b2, acc2);
        acc3 = WMMA_F16(a, b3, acc3);
    }

    int n = lane & 15;
    float part[8];
#pragma unroll
    for (int v = 0; v < 8; ++v) part[v] = 0.0f;
#pragma unroll
    for (int nt = 0; nt < 4; ++nt) {
        v8f acc = (nt == 0) ? acc0 : (nt == 1) ? acc1 : (nt == 2) ? acc2 : acc3;
        float bb = b1[nt * 16 + n];
        float w2 = W2[nt * 16 + n];
#pragma unroll
        for (int v = 0; v < 8; ++v) {
            float h = fmaxf(acc[v] + bb, 0.0f);
            part[v] += h * w2;
        }
    }
    // reduce over the 16 lanes of each half (column dimension of D tile)
#pragma unroll
    for (int mask = 1; mask <= 8; mask <<= 1)
#pragma unroll
        for (int v = 0; v < 8; ++v)
            part[v] += __shfl_xor(part[v], mask, 32);

    if (n == 0) {
        float bb2 = b2[0];
#pragma unroll
        for (int v = 0; v < 8; ++v)
            out[(size_t)tile * 16 + half * 8 + v] = part[v] + bb2;
    }
}

// ---------------------------------------------------------------------------
// launch
// ---------------------------------------------------------------------------
extern "C" void kernel_launch(void* const* d_in, const int* in_sizes, int n_in,
                              void* d_out, int out_size, void* d_ws, size_t ws_size,
                              hipStream_t stream) {
    const int*   edge_index = (const int*)d_in[0];
    const int*   edge_type  = (const int*)d_in[1];
    const int*   edge_pairs = (const int*)d_in[2];
    const float* node_emb   = (const float*)d_in[3];
    const float* comp1      = (const float*)d_in[4];
    const float* basis1     = (const float*)d_in[5];
    const float* root1      = (const float*)d_in[6];
    const float* bias1      = (const float*)d_in[7];
    const float* comp2      = (const float*)d_in[8];
    const float* basis2     = (const float*)d_in[9];
    const float* root2      = (const float*)d_in[10];
    const float* bias2      = (const float*)d_in[11];
    const float* W1         = (const float*)d_in[12];
    const float* b1         = (const float*)d_in[13];
    const float* W2         = (const float*)d_in[14];
    const float* b2         = (const float*)d_in[15];
    float* out = (float*)d_out;

    char* ws = (char*)d_ws;
    float*    sums = (float*)(ws);                    // 204,800,000 B (N*R*D f32)
    float*    cnt  = (float*)(ws + 204800000);        //   3,200,000 B (N*R f32)
    float*    x1   = (float*)(ws + 208000000);        //  25,600,000 B (N*D f32)
    float*    zbuf = (float*)(ws + 233600000);        //  25,600,000 B (N*D f32)
    _Float16* wf1  = (_Float16*)(ws + 259200000);     //      73,728 B
    _Float16* wf2  = (_Float16*)(ws + 259273728);     //      73,728 B
    _Float16* wfm  = (_Float16*)(ws + 259347456);     //      16,384 B

    // pack weights into WMMA fragment layout (tiny)
    pack_rgcn_w<<<9, 256, 0, stream>>>(comp1, basis1, root1, wf1);
    pack_rgcn_w<<<9, 256, 0, stream>>>(comp2, basis2, root2, wf2);
    pack_mlp_w<<<2, 256, 0, stream>>>(W1, wfm);

    const long n4 = 13000000L;  // (204.8MB + 3.2MB) / 16

    // layer 1
    zero_kernel<<<2048, 256, 0, stream>>>(sums, n4);
    scatter_kernel<<<125000, 256, 0, stream>>>(edge_index, edge_type, node_emb, sums, cnt);
    rgcn_gemm<<<3125, 64, 0, stream>>>(sums, cnt, node_emb, wf1, bias1, x1, 1);

    // layer 2
    zero_kernel<<<2048, 256, 0, stream>>>(sums, n4);
    scatter_kernel<<<125000, 256, 0, stream>>>(edge_index, edge_type, x1, sums, cnt);
    rgcn_gemm<<<3125, 64, 0, stream>>>(sums, cnt, x1, wf2, bias2, zbuf, 0);

    // decoder
    decode_kernel<<<15625, 64, 0, stream>>>(edge_pairs, zbuf, wfm, b1, W2, b2, out);
}
